// MemoryGnn_36155034697859
// MI455X (gfx1250) — matrix operand order repaired and checked by reading, not compiled, not run on hardware
//
#include <hip/hip_runtime.h>
#include <math.h>

typedef _Float16 h16;
typedef __attribute__((ext_vector_type(16))) _Float16 v16h;
typedef __attribute__((ext_vector_type(8)))  _Float16 v8h;
typedef __attribute__((ext_vector_type(8)))  float    v8f;

#define DEV static __device__ __forceinline__

DEV float wred_sum(float v) {
#pragma unroll
  for (int o = 16; o; o >>= 1) v += __shfl_xor(v, o, 32);
  return v;
}

// ---------------------------------------------------------------------------
// Generic WMMA GEMM:  C[P=16384][Nn] = A[P][K](f16) * Bw[Npad][K](f16,T) + bias
// Bw is zero-padded to a multiple of 128 rows -> no masking in the hot loop.
// wave tile 32(M) x 64(N); block = 8 waves as 4(M) x 2(N) -> 128x128 tile.
// mode 0: Cf = acc+bias | 1: Cf += acc+bias | 2: Ch = relu f16 | 3: Ch = gelu f16
// ---------------------------------------------------------------------------
__global__ __launch_bounds__(256)
void gemm_kernel(const h16* __restrict__ A, const h16* __restrict__ Bw,
                 const float* __restrict__ bias, float* __restrict__ Cf,
                 h16* __restrict__ Ch, int Nn, int K, int mode) {
  const int lane = threadIdx.x & 31;
  const int wave = threadIdx.x >> 5;
  const int lr = lane & 15;           // row/col within 16-tile
  const int hi = lane >> 4;           // lane-half select
  const int wm = wave & 3;            // wave position in M
  const int wn = wave >> 2;           // wave position in N
  const int m0 = blockIdx.x * 128 + wm * 32;
  const int n0 = blockIdx.y * 128 + wn * 64;

  // A fragment rows (two 16-row tiles); 8*hi folded into base
  const h16* arow0 = A + (size_t)(m0 + lr) * K + 8 * hi;
  const h16* arow1 = arow0 + (size_t)16 * K;
  // B fragment rows (always valid: weights zero-padded)
  const h16* brow[4];
#pragma unroll
  for (int t = 0; t < 4; ++t)
    brow[t] = Bw + (size_t)(n0 + t * 16 + lr) * K + 8 * hi;

  v8f acc[2][4] = {};
  for (int k0 = 0; k0 < K; k0 += 32) {
    v16h a0, a1;
    {
      v8h lo0 = *(const v8h*)(arow0 + k0);
      v8h hh0 = *(const v8h*)(arow0 + k0 + 16);
      v8h lo1 = *(const v8h*)(arow1 + k0);
      v8h hh1 = *(const v8h*)(arow1 + k0 + 16);
#pragma unroll
      for (int j = 0; j < 8; ++j) {
        a0[j] = lo0[j]; a0[j + 8] = hh0[j];
        a1[j] = lo1[j]; a1[j + 8] = hh1[j];
      }
    }
#pragma unroll
    for (int t = 0; t < 4; ++t) {
      v8h b0 = *(const v8h*)(brow[t] + k0);
      v8h b1 = *(const v8h*)(brow[t] + k0 + 16);
      v16h b;
#pragma unroll
      for (int j = 0; j < 8; ++j) { b[j] = b0[j]; b[j + 8] = b1[j]; }
      acc[0][t] = __builtin_amdgcn_wmma_f32_16x16x32_f16(
          false, a0, false, b, (short)0, acc[0][t], false, false);
      acc[1][t] = __builtin_amdgcn_wmma_f32_16x16x32_f16(
          false, a1, false, b, (short)0, acc[1][t], false, false);
    }
  }
  // epilogue: C/D layout -> row = m0 + s*16 + r + 8*hi, col = n0 + t*16 + lr
#pragma unroll
  for (int t = 0; t < 4; ++t) {
    int cn = n0 + t * 16 + lr;
    if (cn >= Nn) continue;
    float bb = bias ? bias[cn] : 0.0f;
#pragma unroll
    for (int s = 0; s < 2; ++s) {
#pragma unroll
      for (int r = 0; r < 8; ++r) {
        int row = m0 + s * 16 + r + 8 * hi;
        float v = acc[s][t][r] + bb;
        size_t o = (size_t)row * Nn + cn;
        if (mode == 0) {
          Cf[o] = v;
        } else if (mode == 1) {
          Cf[o] += v;
        } else if (mode == 2) {
          Ch[o] = (h16)fmaxf(v, 0.0f);
        } else {
          float u = 0.7978845608f * (v + 0.044715f * v * v * v);
          Ch[o] = (h16)(0.5f * v * (1.0f + tanhf(u)));
        }
      }
    }
  }
}

// ---------------------------------------------------------------------------
// gsf[b][c] = max_k sa4[b][c][k]   (one wave per (b,c))
// ---------------------------------------------------------------------------
__global__ __launch_bounds__(256)
void gsf_kernel(const float* __restrict__ sa4, float* __restrict__ gsf) {
  int idx = blockIdx.x * 8 + (threadIdx.x >> 5);
  int lane = threadIdx.x & 31;
  const float* row = sa4 + (size_t)idx * 256;
  float m = -3.4e38f;
#pragma unroll
  for (int j = 0; j < 8; ++j) m = fmaxf(m, row[lane + j * 32]);
#pragma unroll
  for (int o = 16; o; o >>= 1) m = fmaxf(m, __shfl_xor(m, o, 32));
  if (lane == 0) gsf[idx] = m;
}

// ---------------------------------------------------------------------------
// Gather + fuse + frame-pairing -> A0[p=e*1024+n][c=0..2047] f16
// ---------------------------------------------------------------------------
__global__ __launch_bounds__(256)
void fuse_kernel(const float* __restrict__ pose, const float* __restrict__ grasp,
                 const float* __restrict__ color, const float* __restrict__ seed,
                 const float* __restrict__ gsf, const int* __restrict__ idxs,
                 h16* __restrict__ A0) {
  int p = blockIdx.x;
  int c = blockIdx.y * 256 + threadIdx.x;
  int e = p >> 10, n = p & 1023;
  int g = e >> 2, t = (e & 3) + 1;
  int b = g * 5 + ((c < 1024) ? 0 : t);
  int cc = c & 1023;
  int id = idxs[b * 1024 + n];
  float v = pose[((size_t)b * 1024 + cc) * 2048 + id];
  if (cc < 256)      v += grasp[((size_t)b * 256 + cc) * 2048 + id];
  else if (cc < 512) v += color[((size_t)b * 256 + (cc - 256)) * 2048 + id];
  else if (cc < 768) v += seed[((size_t)b * 256 + (cc - 512)) * 2048 + id];
  else               v += gsf[b * 256 + (cc - 768)];
  A0[(size_t)p * 2048 + c] = (h16)v;
}

// ---------------------------------------------------------------------------
// Train-mode BatchNorm over P=16384 rows per channel + ReLU -> f16
// block = 64 channels x 4 row-lanes, deterministic two-pass
// ---------------------------------------------------------------------------
__global__ __launch_bounds__(256)
void bn_relu_kernel(const float* __restrict__ y, h16* __restrict__ out,
                    const float* __restrict__ g, const float* __restrict__ b, int C) {
  const int P = 16384;
  int cx = threadIdx.x & 63, ry = threadIdx.x >> 6;
  int c = blockIdx.x * 64 + cx;
  float s = 0.f, s2 = 0.f;
  for (int r = ry; r < P; r += 4) {
    float v = y[(size_t)r * C + c];
    s += v; s2 += v * v;
  }
  __shared__ float shs[4][64], shq[4][64], shm[64], shr[64];
  shs[ry][cx] = s; shq[ry][cx] = s2;
  __syncthreads();
  if (ry == 0) {
    float ts = shs[0][cx] + shs[1][cx] + shs[2][cx] + shs[3][cx];
    float tq = shq[0][cx] + shq[1][cx] + shq[2][cx] + shq[3][cx];
    float m = ts * (1.0f / P);
    float var = tq * (1.0f / P) - m * m;
    shm[cx] = m; shr[cx] = rsqrtf(var + 1e-5f);
  }
  __syncthreads();
  float scale = g[c] * shr[cx];
  float shift = b[c] - shm[cx] * scale;
  for (int r = ry; r < P; r += 4) {
    float v = y[(size_t)r * C + c] * scale + shift;
    out[(size_t)r * C + c] = (h16)fmaxf(v, 0.0f);
  }
}

// ---------------------------------------------------------------------------
// Reshape conv output to token stream + add sinusoidal PE
// tok = (g*1024+n)*4 + f  ;  src point = (g*4+f)*1024 + n
// ---------------------------------------------------------------------------
__global__ __launch_bounds__(256)
void build_tokens(const float* __restrict__ y4, float* __restrict__ x) {
  int tok = blockIdx.x;
  int f = tok & 3;
  int sg = tok >> 2;
  int g = sg >> 10, n = sg & 1023;
  const float* src = y4 + ((size_t)(g * 4 + f) * 1024 + n) * 512;
  float* dst = x + (size_t)tok * 512;
  for (int d = threadIdx.x; d < 512; d += 256) {
    int i = d >> 1;
    float fr = expf(-9.210340371976184f * (float)(2 * i) * (1.0f / 512.0f));
    float pe = (d & 1) ? cosf((float)f * fr) : sinf((float)f * fr);
    dst[d] = src[d] + pe;
  }
}

// ---------------------------------------------------------------------------
// LayerNorm over 512 dims, one wave per token, writes f16
// ---------------------------------------------------------------------------
__global__ __launch_bounds__(256)
void ln_kernel(const float* __restrict__ x, h16* __restrict__ out,
               const float* __restrict__ g, const float* __restrict__ b) {
  int tok = blockIdx.x * 8 + (threadIdx.x >> 5);
  int lane = threadIdx.x & 31;
  const float* row = x + (size_t)tok * 512;
  float4 v[4];
  float s = 0.f, s2 = 0.f;
#pragma unroll
  for (int j = 0; j < 4; ++j) {
    v[j] = *(const float4*)(row + lane * 16 + j * 4);
    s += v[j].x + v[j].y + v[j].z + v[j].w;
    s2 += v[j].x * v[j].x + v[j].y * v[j].y + v[j].z * v[j].z + v[j].w * v[j].w;
  }
  s = wred_sum(s); s2 = wred_sum(s2);
  float m = s * (1.0f / 512.0f);
  float var = s2 * (1.0f / 512.0f) - m * m;
  float rs = rsqrtf(var + 1e-5f);
  h16* orow = out + (size_t)tok * 512;
#pragma unroll
  for (int j = 0; j < 4; ++j) {
    int d = lane * 16 + j * 4;
    float4 gg = *(const float4*)(g + d);
    float4 bb = *(const float4*)(b + d);
    orow[d + 0] = (h16)((v[j].x - m) * rs * gg.x + bb.x);
    orow[d + 1] = (h16)((v[j].y - m) * rs * gg.y + bb.y);
    orow[d + 2] = (h16)((v[j].z - m) * rs * gg.z + bb.z);
    orow[d + 3] = (h16)((v[j].w - m) * rs * gg.w + bb.w);
  }
}

// ---------------------------------------------------------------------------
// Attention: L=4, heads=8, dh=64. One wave per (seq, head); lane owns 2 dims.
// ---------------------------------------------------------------------------
__global__ __launch_bounds__(256)
void attn_kernel(const float* __restrict__ qkv, h16* __restrict__ attn) {
  int head = threadIdx.x >> 5, lane = threadIdx.x & 31;
  size_t seq = blockIdx.x;
  int d0 = head * 64 + lane * 2;
  float q[4][2], k[4][2], v[4][2];
#pragma unroll
  for (int t = 0; t < 4; ++t) {
    const float* row = qkv + (seq * 4 + t) * 1536;
    q[t][0] = row[d0];        q[t][1] = row[d0 + 1];
    k[t][0] = row[512 + d0];  k[t][1] = row[512 + d0 + 1];
    v[t][0] = row[1024 + d0]; v[t][1] = row[1024 + d0 + 1];
  }
  float s[4][4];
#pragma unroll
  for (int i = 0; i < 4; ++i)
#pragma unroll
    for (int j = 0; j < 4; ++j) {
      float p = q[i][0] * k[j][0] + q[i][1] * k[j][1];
      s[i][j] = wred_sum(p) * 0.125f;  // 1/sqrt(64)
    }
#pragma unroll
  for (int i = 0; i < 4; ++i) {
    float mx = fmaxf(fmaxf(s[i][0], s[i][1]), fmaxf(s[i][2], s[i][3]));
    float e0 = expf(s[i][0] - mx), e1 = expf(s[i][1] - mx);
    float e2 = expf(s[i][2] - mx), e3 = expf(s[i][3] - mx);
    float inv = 1.0f / (e0 + e1 + e2 + e3);
    float o0 = (e0 * v[0][0] + e1 * v[1][0] + e2 * v[2][0] + e3 * v[3][0]) * inv;
    float o1 = (e0 * v[0][1] + e1 * v[1][1] + e2 * v[2][1] + e3 * v[3][1]) * inv;
    h16* orow = attn + (seq * 4 + i) * 512;
    orow[d0] = (h16)o0; orow[d0 + 1] = (h16)o1;
  }
}

// ---------------------------------------------------------------------------
// Weight prep: cast & transpose-cast to f16 (transpose zero-pads rows to Npad)
// ---------------------------------------------------------------------------
__global__ __launch_bounds__(256)
void cast_kernel(const float* __restrict__ s, h16* __restrict__ d, int n) {
  int i = blockIdx.x * 256 + threadIdx.x;
  if (i < n) d[i] = (h16)s[i];
}

__global__ __launch_bounds__(256)
void transpose_kernel(const float* __restrict__ s, h16* __restrict__ d,
                      int K, int N, int Npad) {
  size_t i = (size_t)blockIdx.x * 256 + threadIdx.x;   // i = n*K + k
  if (i >= (size_t)Npad * K) return;
  int n = (int)(i / K), k = (int)(i % K);
  d[i] = (n < N) ? (h16)s[(size_t)k * N + n] : (h16)0;
}

// ---------------------------------------------------------------------------
// Final: rot6d (Gram-Schmidt) + concat [trans(3), rot(9)]
// ---------------------------------------------------------------------------
__global__ __launch_bounds__(256)
void head_out(const float* __restrict__ t3, const float* __restrict__ r3,
              float* __restrict__ out) {
  int tok = blockIdx.x * 256 + threadIdx.x;
  const float* d6 = r3 + (size_t)tok * 6;
  float a1x = d6[0], a1y = d6[1], a1z = d6[2];
  float a2x = d6[3], a2y = d6[4], a2z = d6[5];
  float n1 = fmaxf(sqrtf(a1x * a1x + a1y * a1y + a1z * a1z), 1e-12f);
  float b1x = a1x / n1, b1y = a1y / n1, b1z = a1z / n1;
  float dt = b1x * a2x + b1y * a2y + b1z * a2z;
  float ux = a2x - dt * b1x, uy = a2y - dt * b1y, uz = a2z - dt * b1z;
  float n2 = fmaxf(sqrtf(ux * ux + uy * uy + uz * uz), 1e-12f);
  float b2x = ux / n2, b2y = uy / n2, b2z = uz / n2;
  float b3x = b1y * b2z - b1z * b2y;
  float b3y = b1z * b2x - b1x * b2z;
  float b3z = b1x * b2y - b1y * b2x;
  float* o = out + (size_t)tok * 12;
  o[0] = t3[tok * 3 + 0]; o[1] = t3[tok * 3 + 1]; o[2] = t3[tok * 3 + 2];
  o[3] = b1x; o[4] = b1y; o[5] = b1z;
  o[6] = b2x; o[7] = b2y; o[8] = b2z;
  o[9] = b3x; o[10] = b3y; o[11] = b3z;
}

// ---------------------------------------------------------------------------
extern "C" void kernel_launch(void* const* d_in, const int* in_sizes, int n_in,
                              void* d_out, int out_size, void* d_ws, size_t ws_size,
                              hipStream_t stream) {
  const float* fp2    = (const float*)d_in[0];
  const float* lgrasp = (const float*)d_in[1];
  const float* lcolor = (const float*)d_in[2];
  const float* pose   = (const float*)d_in[3];
  const float* sa4    = (const float*)d_in[6];
  const int*   sidx   = (const int*)d_in[7];
  const float* c1w = (const float*)d_in[8],  *c1b = (const float*)d_in[9];
  const float* c2w = (const float*)d_in[10], *c2b = (const float*)d_in[11];
  const float* c3w = (const float*)d_in[12], *c3b = (const float*)d_in[13];
  const float* c4w = (const float*)d_in[14], *c4b = (const float*)d_in[15];
  const float* bn1g = (const float*)d_in[16], *bn1b = (const float*)d_in[17];
  const float* bn2g = (const float*)d_in[18], *bn2b = (const float*)d_in[19];
  const float* bn3g = (const float*)d_in[20], *bn3b = (const float*)d_in[21];
  const float* ln1g = (const float*)d_in[22], *ln1b = (const float*)d_in[23];
  const float* qkvw = (const float*)d_in[24], *qkvb = (const float*)d_in[25];
  const float* outw = (const float*)d_in[26], *outb = (const float*)d_in[27];
  const float* ln2g = (const float*)d_in[28], *ln2b = (const float*)d_in[29];
  const float* m1w = (const float*)d_in[30], *m1b = (const float*)d_in[31];
  const float* m2w = (const float*)d_in[32], *m2b = (const float*)d_in[33];
  const float* lnfg = (const float*)d_in[34], *lnfb = (const float*)d_in[35];
  const float* projw = (const float*)d_in[36], *projb = (const float*)d_in[37];
  const float* rw1 = (const float*)d_in[38], *rb1 = (const float*)d_in[39];
  const float* rw2 = (const float*)d_in[40], *rb2 = (const float*)d_in[41];
  const float* rw3 = (const float*)d_in[42], *rb3 = (const float*)d_in[43];
  const float* tw1 = (const float*)d_in[44], *tb1 = (const float*)d_in[45];
  const float* tw2 = (const float*)d_in[46], *tb2 = (const float*)d_in[47];
  const float* tw3 = (const float*)d_in[48], *tb3 = (const float*)d_in[49];

  // ---- workspace arena (bump allocated, deterministic) ----
  char* wsb = (char*)d_ws;
  size_t cur = 0;
  auto alloc = [&](size_t bytes) -> void* {
    void* p = wsb + cur;
    cur += (bytes + 255) & ~(size_t)255;
    return p;
  };
  h16* Wc1  = (h16*)alloc((size_t)1536 * 2048 * 2);
  h16* Wc2  = (h16*)alloc((size_t)1024 * 1536 * 2);
  h16* Wc3  = (h16*)alloc((size_t)768 * 1024 * 2);
  h16* Wc4  = (h16*)alloc((size_t)512 * 768 * 2);
  h16* Wqkv = (h16*)alloc((size_t)12 * 1536 * 512 * 2);
  h16* Wout = (h16*)alloc((size_t)12 * 512 * 512 * 2);
  h16* Wm1  = (h16*)alloc((size_t)12 * 512 * 512 * 2);
  h16* Wm2  = (h16*)alloc((size_t)12 * 512 * 512 * 2);
  h16* Wproj = (h16*)alloc((size_t)512 * 512 * 2);
  h16* Wr1 = (h16*)alloc((size_t)256 * 512 * 2);
  h16* Wr2 = (h16*)alloc((size_t)128 * 256 * 2);
  h16* Wr3 = (h16*)alloc((size_t)128 * 128 * 2);   // 6 rows padded to 128
  h16* Wt1 = (h16*)alloc((size_t)256 * 512 * 2);
  h16* Wt2 = (h16*)alloc((size_t)128 * 256 * 2);
  h16* Wt3 = (h16*)alloc((size_t)128 * 128 * 2);   // 3 rows padded to 128
  float* GSF = (float*)alloc(5120 * 4);
  h16*  A0  = (h16*)alloc((size_t)16384 * 2048 * 2);
  float* Y  = (float*)alloc((size_t)16384 * 1536 * 4);
  h16*  ACT = (h16*)alloc((size_t)16384 * 1536 * 2);
  float* X  = (float*)alloc((size_t)16384 * 512 * 4);
  float* R3 = (float*)alloc((size_t)16384 * 6 * 4);
  float* T3 = (float*)alloc((size_t)16384 * 3 * 4);
  h16* slotA = ACT;
  h16* slotB = ACT + (size_t)16384 * 512;
  h16* Hmid1 = ACT + (size_t)2 * 16384 * 512;        // 16384x256 f16
  h16* Hmid2 = Hmid1 + (size_t)16384 * 256;          // 16384x128 f16

  auto cast = [&](const float* s, h16* d, size_t n) {
    cast_kernel<<<(unsigned)((n + 255) / 256), 256, 0, stream>>>(s, d, (int)n);
  };
  auto tpose = [&](const float* s, h16* d, int K, int N, int Npad) {
    transpose_kernel<<<(unsigned)(((size_t)Npad * K + 255) / 256), 256, 0, stream>>>(
        s, d, K, N, Npad);
  };
  auto gemm = [&](const h16* A, const h16* Bw, const float* bias, float* Cf,
                  h16* Ch, int Nn, int K, int mode) {
    dim3 grid(16384 / 128, (Nn + 127) / 128);
    gemm_kernel<<<grid, 256, 0, stream>>>(A, Bw, bias, Cf, Ch, Nn, K, mode);
  };
  auto ln = [&](const float* xin, h16* hout, const float* g, const float* b) {
    ln_kernel<<<2048, 256, 0, stream>>>(xin, hout, g, b);
  };

  // ---- weight prep (conv weights are already [out][in]; others transposed) ----
  cast(c1w, Wc1, (size_t)1536 * 2048);
  cast(c2w, Wc2, (size_t)1024 * 1536);
  cast(c3w, Wc3, (size_t)768 * 1024);
  cast(c4w, Wc4, (size_t)512 * 768);
  for (int l = 0; l < 12; ++l) {
    tpose(qkvw + (size_t)l * 512 * 1536, Wqkv + (size_t)l * 1536 * 512, 512, 1536, 1536);
    tpose(outw + (size_t)l * 512 * 512,  Wout + (size_t)l * 512 * 512, 512, 512, 512);
    tpose(m1w  + (size_t)l * 512 * 512,  Wm1  + (size_t)l * 512 * 512, 512, 512, 512);
    tpose(m2w  + (size_t)l * 512 * 512,  Wm2  + (size_t)l * 512 * 512, 512, 512, 512);
  }
  tpose(projw, Wproj, 512, 512, 512);
  tpose(rw1, Wr1, 512, 256, 256);
  tpose(rw2, Wr2, 256, 128, 128);
  tpose(rw3, Wr3, 128, 6, 128);
  tpose(tw1, Wt1, 512, 256, 256);
  tpose(tw2, Wt2, 256, 128, 128);
  tpose(tw3, Wt3, 128, 3, 128);

  // ---- gather/fuse front end ----
  gsf_kernel<<<640, 256, 0, stream>>>(sa4, GSF);
  fuse_kernel<<<dim3(16384, 8), 256, 0, stream>>>(pose, lgrasp, lcolor, fp2, GSF, sidx, A0);

  // ---- conv stack (GEMM + BN + ReLU) ----
  gemm(A0, Wc1, c1b, Y, nullptr, 1536, 2048, 0);
  bn_relu_kernel<<<1536 / 64, 256, 0, stream>>>(Y, ACT, bn1g, bn1b, 1536);
  gemm(ACT, Wc2, c2b, Y, nullptr, 1024, 1536, 0);
  bn_relu_kernel<<<1024 / 64, 256, 0, stream>>>(Y, ACT, bn2g, bn2b, 1024);
  gemm(ACT, Wc3, c3b, Y, nullptr, 768, 1024, 0);
  bn_relu_kernel<<<768 / 64, 256, 0, stream>>>(Y, ACT, bn3g, bn3b, 768);
  gemm(ACT, Wc4, c4b, Y, nullptr, 512, 768, 0);

  // ---- tokens + positional encoding ----
  build_tokens<<<16384, 256, 0, stream>>>(Y, X);

  // ---- 12 transformer layers ----
  for (int l = 0; l < 12; ++l) {
    ln(X, slotA, ln1g + l * 512, ln1b + l * 512);
    gemm(slotA, Wqkv + (size_t)l * 1536 * 512, qkvb + l * 1536, Y, nullptr, 1536, 512, 0);
    attn_kernel<<<4096, 256, 0, stream>>>(Y, slotB);
    gemm(slotB, Wout + (size_t)l * 512 * 512, outb + l * 512, X, nullptr, 512, 512, 1);
    ln(X, slotA, ln2g + l * 512, ln2b + l * 512);
    gemm(slotA, Wm1 + (size_t)l * 512 * 512, m1b + l * 512, nullptr, slotB, 512, 512, 3);
    gemm(slotB, Wm2 + (size_t)l * 512 * 512, m2b + l * 512, X, nullptr, 512, 512, 1);
  }

  // ---- final LN + projection ----
  ln(X, slotA, lnfg, lnfb);
  gemm(slotA, Wproj, projb, Y, nullptr, 512, 512, 0);
  cast(Y, slotB, (size_t)16384 * 512);

  // ---- rotation head ----
  gemm(slotB, Wr1, rb1, nullptr, Hmid1, 256, 512, 2);
  gemm(Hmid1, Wr2, rb2, nullptr, Hmid2, 128, 256, 2);
  gemm(Hmid2, Wr3, rb3, R3, nullptr, 6, 128, 0);
  // ---- translation head (reuses Hmid buffers) ----
  gemm(slotB, Wt1, tb1, nullptr, Hmid1, 256, 512, 2);
  gemm(Hmid1, Wt2, tb2, nullptr, Hmid2, 128, 256, 2);
  gemm(Hmid2, Wt3, tb3, T3, nullptr, 3, 128, 0);

  // ---- rot6d + assemble output (4096, 4, 12) f32 ----
  head_out<<<64, 256, 0, stream>>>(T3, R3, (float*)d_out);
}
